// ResnetBlocWithAttn_8057358648213
// MI455X (gfx1250) — compile-verified, matrix operand order checked by reading, not run on hardware
//
#include <hip/hip_runtime.h>
#include <hip/hip_bf16.h>

typedef __attribute__((ext_vector_type(16))) _Float16 v16h;
typedef __attribute__((ext_vector_type(8)))  _Float16 v8h;
typedef __attribute__((ext_vector_type(8)))  float    v8f;
typedef __attribute__((ext_vector_type(4)))  int      vi4;

#define WMMA_F16(A,B,Cacc) __builtin_amdgcn_wmma_f32_16x16x32_f16(false,(A),false,(B),(short)0,(Cacc),false,false)

#if defined(__HIP_DEVICE_COMPILE__) && __has_builtin(__builtin_amdgcn_global_load_async_to_lds_b128)
#define HAVE_ASYNC_LDS 1
#else
#define HAVE_ASYNC_LDS 0
#endif

static __device__ __forceinline__ v16h cat8(v8h lo, v8h hi) {
  return __builtin_shufflevector(lo, hi, 0,1,2,3,4,5,6,7,8,9,10,11,12,13,14,15);
}

// ---------------- constants ----------------
// B=4, C=64, H=W=D=16, N=4096, groups=16 (4 ch/group), TEMB=256

// ---------------- GroupNorm stats ----------------
__global__ __launch_bounds__(256) void gn_stats_cf(const float* __restrict__ x,
                                                   float* __restrict__ stats) {
  int bg = blockIdx.x; int b = bg >> 4, g = bg & 15;
  const float* base = x + (((size_t)b * 64) + g * 4) * 4096;
  float s = 0.f, s2 = 0.f;
  for (int i = threadIdx.x; i < 4 * 4096; i += 256) { float v = base[i]; s += v; s2 += v * v; }
  __shared__ float rs[256], rq[256];
  rs[threadIdx.x] = s; rq[threadIdx.x] = s2; __syncthreads();
  for (int st = 128; st > 0; st >>= 1) {
    if ((int)threadIdx.x < st) { rs[threadIdx.x] += rs[threadIdx.x + st]; rq[threadIdx.x] += rq[threadIdx.x + st]; }
    __syncthreads();
  }
  if (threadIdx.x == 0) {
    float mu = rs[0] / 16384.f;
    float var = rq[0] / 16384.f - mu * mu;
    stats[bg * 2] = mu; stats[bg * 2 + 1] = rsqrtf(var + 1e-5f);
  }
}

__global__ __launch_bounds__(256) void gn_stats_cl(const float* __restrict__ h,
                                                   float* __restrict__ stats) {
  int bg = blockIdx.x; int b = bg >> 4, g = bg & 15;
  float s = 0.f, s2 = 0.f;
  for (int i = threadIdx.x; i < 4 * 4096; i += 256) {
    int n = i >> 2, c = i & 3;
    float v = h[(((size_t)b * 4096) + n) * 64 + g * 4 + c];
    s += v; s2 += v * v;
  }
  __shared__ float rs[256], rq[256];
  rs[threadIdx.x] = s; rq[threadIdx.x] = s2; __syncthreads();
  for (int st = 128; st > 0; st >>= 1) {
    if ((int)threadIdx.x < st) { rs[threadIdx.x] += rs[threadIdx.x + st]; rq[threadIdx.x] += rq[threadIdx.x + st]; }
    __syncthreads();
  }
  if (threadIdx.x == 0) {
    float mu = rs[0] / 16384.f;
    float var = rq[0] / 16384.f - mu * mu;
    stats[bg * 2] = mu; stats[bg * 2 + 1] = rsqrtf(var + 1e-5f);
  }
}

// ---------------- GroupNorm apply (+optional swish), emit f16 channels-last ----------------
__global__ __launch_bounds__(256) void gn_apply_cf(const float* __restrict__ x,
                                                   const float* __restrict__ gamma,
                                                   const float* __restrict__ beta,
                                                   const float* __restrict__ stats,
                                                   _Float16* __restrict__ act, int do_swish) {
  int tid = blockIdx.x * 256 + threadIdx.x;      // [b][n][c], c fastest
  int c = tid & 63; int n = (tid >> 6) & 4095; int b = tid >> 18;
  int g = c >> 2;
  float mu = stats[(b * 16 + g) * 2], rstd = stats[(b * 16 + g) * 2 + 1];
  float v = x[(((size_t)b * 64) + c) * 4096 + n];
  v = (v - mu) * rstd * gamma[c] + beta[c];
  if (do_swish) v = v / (1.f + __expf(-v));
  act[tid] = (_Float16)v;
}

__global__ __launch_bounds__(256) void gn_apply_cl(const float* __restrict__ h,
                                                   const float* __restrict__ gamma,
                                                   const float* __restrict__ beta,
                                                   const float* __restrict__ stats,
                                                   _Float16* __restrict__ act, int do_swish) {
  int tid = blockIdx.x * 256 + threadIdx.x;
  int c = tid & 63; int b = tid >> 18;
  int g = c >> 2;
  float mu = stats[(b * 16 + g) * 2], rstd = stats[(b * 16 + g) * 2 + 1];
  float v = h[tid];
  v = (v - mu) * rstd * gamma[c] + beta[c];
  if (do_swish) v = v / (1.f + __expf(-v));
  act[tid] = (_Float16)v;
}

// ---------------- time embedding MLP (tiny: 4x256 @ 256x64) ----------------
__global__ __launch_bounds__(256) void time_mlp_k(const float* __restrict__ temb,
                                                  const float* __restrict__ w,
                                                  const float* __restrict__ bias,
                                                  float* __restrict__ t) {
  int tid = threadIdx.x; int b = tid >> 6, c = tid & 63;
  float acc = bias[c];
  for (int k = 0; k < 256; ++k) {
    float e = temb[b * 256 + k];
    e = e / (1.f + __expf(-e));
    acc += e * w[c * 256 + k];
  }
  t[tid] = acc;
}

// ---------------- weight packing into WMMA B-fragment order ----------------
// B-frag (32x16 f16): lane l holds col n=l&15; elems 0..7 -> K=(l>>4)*8+e, 8..15 -> K=16+(l>>4)*8+(e-8)
__global__ __launch_bounds__(256) void pack_conv_k(const float* __restrict__ w,
                                                   _Float16* __restrict__ dst) {
  int tid = blockIdx.x * 256 + threadIdx.x;
  if (tid >= 4 * 27 * 2 * 32 * 16) return;
  int e = tid & 15; int lane = (tid >> 4) & 31; int kk = (tid >> 9) & 1;
  int tap = (tid >> 10) % 27; int cot = tid / 27648;
  int hi = lane >> 4;
  int k = (e < 8) ? (hi * 8 + e) : (16 + hi * 8 + (e - 8));
  int ci = kk * 32 + k;
  int co = cot * 16 + (lane & 15);
  dst[tid] = (_Float16)w[(((size_t)co * 64) + ci) * 27 + tap];
}

__global__ __launch_bounds__(256) void pack_mat_k(const float* __restrict__ w,
                                                  _Float16* __restrict__ dst, int ntiles) {
  int tid = blockIdx.x * 256 + threadIdx.x;
  if (tid >= ntiles * 1024) return;
  int e = tid & 15; int lane = (tid >> 4) & 31; int kk = (tid >> 9) & 1; int ot = tid >> 10;
  int hi = lane >> 4;
  int k = (e < 8) ? (hi * 8 + e) : (16 + hi * 8 + (e - 8));
  int ci = kk * 32 + k;
  int o = ot * 16 + (lane & 15);
  dst[tid] = (_Float16)w[o * 64 + ci];
}

// ---------------- Conv3d 3x3x3 as implicit GEMM via WMMA ----------------
// act: f16 channels-last [b][n][c]; out: f32 channels-last [b][n][co]
// block = 128 thr = 4 waves (one co-tile each); grid = B*256 (one (b,h,w) column, 16 d-positions)
__global__ __launch_bounds__(128) void conv3_wmma(const _Float16* __restrict__ act,
                                                  const _Float16* __restrict__ wpack,
                                                  const float* __restrict__ bias_c,
                                                  const float* __restrict__ bias_bc,
                                                  const float* __restrict__ resid_cf,
                                                  float* __restrict__ out_cl) {
  __shared__ _Float16 smem[3 * 3 * 18 * 64];   // halo: 3h x 3w x (16+2)d x 64ci f16 = 20.25 KB
  int b = blockIdx.x >> 8; int hw = blockIdx.x & 255;
  int h0 = hw >> 4, w0 = hw & 15;

  // stage halo (zero-padded): CDNA5 async memory->LDS DMA when available
#if HAVE_ASYNC_LDS
  {
    typedef __attribute__((address_space(3))) vi4 lds_vi4;
    for (int i = threadIdx.x; i < 1296; i += 128) {
      int ci8 = (i & 7) * 8;
      int id = (i >> 3) % 18;
      int iw = ((i >> 3) / 18) % 3;
      int ih = (i >> 3) / 54;
      int gh = h0 + ih - 1, gw = w0 + iw - 1, gd = id - 1;
      if ((unsigned)gh < 16u && (unsigned)gw < 16u && (unsigned)gd < 16u) {
        const _Float16* gp =
            &act[(((size_t)b * 4096) + ((gh * 16 + gw) * 16 + gd)) * 64 + ci8];
        // generic LDS address carries the LDS byte offset in its low 32 bits
        __builtin_amdgcn_global_load_async_to_lds_b128(
            (vi4*)gp,
            (lds_vi4*)(unsigned long long)(unsigned)(unsigned long long)&smem[i * 8],
            0, 0);
      } else {
        v8h z = {};
        *(v8h*)&smem[i * 8] = z;
      }
    }
#if __has_builtin(__builtin_amdgcn_s_wait_asynccnt)
    __builtin_amdgcn_s_wait_asynccnt(0);
#else
    asm volatile("s_wait_asynccnt 0x0" ::: "memory");
#endif
  }
#else
  for (int i = threadIdx.x; i < 1296; i += 128) {
    int ci8 = (i & 7) * 8;
    int id = (i >> 3) % 18;
    int iw = ((i >> 3) / 18) % 3;
    int ih = (i >> 3) / 54;
    int gh = h0 + ih - 1, gw = w0 + iw - 1, gd = id - 1;
    v8h v = {};
    if ((unsigned)gh < 16u && (unsigned)gw < 16u && (unsigned)gd < 16u)
      v = *(const v8h*)&act[(((size_t)b * 4096) + ((gh * 16 + gw) * 16 + gd)) * 64 + ci8];
    *(v8h*)&smem[i * 8] = v;
  }
#endif
  __syncthreads();

  int lane = threadIdx.x & 31, wave = threadIdx.x >> 5;  // wave = co-tile
  int lo = lane & 15, hi = lane >> 4;
  v8f acc = {};
  for (int tap = 0; tap < 27; ++tap) {
    int ih = tap / 9, iw = (tap / 3) % 3, dz = tap % 3;
    int abase = ((ih * 3 + iw) * 18 + lo + dz) * 64 + hi * 8;
    const _Float16* wp = wpack + (((size_t)(wave * 27 + tap) * 2) * 32 + lane) * 16;
#pragma unroll
    for (int kk = 0; kk < 2; ++kk) {
      v8h alo = *(const v8h*)&smem[abase + kk * 32];
      v8h ahi = *(const v8h*)&smem[abase + kk * 32 + 16];
      v16h A = cat8(alo, ahi);
      v16h Bf = *(const v16h*)(wp + kk * 512);
      acc = WMMA_F16(A, Bf, acc);
    }
  }
  int co = wave * 16 + lo;
  float bc = bias_c[co];
  if (bias_bc) bc += bias_bc[b * 64 + co];
#pragma unroll
  for (int j = 0; j < 8; ++j) {
    int m = j + 8 * hi; int n = hw * 16 + m;
    float v = acc[j] + bc;
    if (resid_cf) v += resid_cf[(((size_t)b * 64) + co) * 4096 + n];
    out_cl[(((size_t)b * 4096) + n) * 64 + co] = v;
  }
}

// ---------------- QKV projection via WMMA ----------------
// q,k -> f16 channels-last [b][n][c] (q pre-scaled 1/8); v -> f16 transposed [b][c][n]
__global__ __launch_bounds__(128) void qkv_wmma(const _Float16* __restrict__ act,
                                                const _Float16* __restrict__ wpack,
                                                _Float16* __restrict__ q_cl,
                                                _Float16* __restrict__ k_cl,
                                                _Float16* __restrict__ v_t) {
  int b = blockIdx.x >> 8; int nt = blockIdx.x & 255; int nbase = nt * 16;
  int lane = threadIdx.x & 31, wave = threadIdx.x >> 5;
  int lo = lane & 15, hi = lane >> 4;
  const _Float16* arow = act + (((size_t)b * 4096) + nbase + lo) * 64 + hi * 8;
  v16h A0 = cat8(*(const v8h*)arow, *(const v8h*)(arow + 16));
  v16h A1 = cat8(*(const v8h*)(arow + 32), *(const v8h*)(arow + 48));
#pragma unroll
  for (int t = 0; t < 3; ++t) {
    int ot = wave + t * 4;                      // 12 o-tiles over 4 waves
    const _Float16* wp = wpack + ((size_t)(ot * 2) * 32 + lane) * 16;
    v8f acc = {};
    acc = WMMA_F16(A0, *(const v16h*)wp, acc);
    acc = WMMA_F16(A1, *(const v16h*)(wp + 512), acc);
    int qi = ot >> 2; int c = (ot * 16 + lo) & 63;
    if (qi == 0) {
#pragma unroll
      for (int j = 0; j < 8; ++j)
        q_cl[(((size_t)b * 4096) + nbase + j + 8 * hi) * 64 + c] = (_Float16)(acc[j] * 0.125f);
    } else if (qi == 1) {
#pragma unroll
      for (int j = 0; j < 8; ++j)
        k_cl[(((size_t)b * 4096) + nbase + j + 8 * hi) * 64 + c] = (_Float16)acc[j];
    } else {
      v8h vv;
#pragma unroll
      for (int j = 0; j < 8; ++j) vv[j] = (_Float16)acc[j];
      *(v8h*)&v_t[(((size_t)b * 64) + c) * 4096 + nbase + 8 * hi] = vv;
    }
  }
}

// ---------------- Flash attention + out-projection + residual ----------------
// 1 wave per 16-row Q tile; 8 waves/block; fuses out_w, out_b, hres residual; stores channels-first
__global__ __launch_bounds__(256) void flash_wmma(const _Float16* __restrict__ q_cl,
                                                  const _Float16* __restrict__ k_cl,
                                                  const _Float16* __restrict__ v_t,
                                                  const _Float16* __restrict__ wpack_o,
                                                  const float* __restrict__ out_b,
                                                  const float* __restrict__ hres,
                                                  float* __restrict__ out_cf) {
  __shared__ _Float16 sws[8 * 1024];
  int wave = threadIdx.x >> 5, lane = threadIdx.x & 31;
  int lo = lane & 15, hi = lane >> 4;
  _Float16* sP = sws + wave * 1024;
  int gw = blockIdx.x * 8 + wave;
  int b = gw >> 8; int nbase = (gw & 255) * 16;

  const _Float16* qrow = q_cl + (((size_t)b * 4096) + nbase + lo) * 64 + hi * 8;
  v16h AQ0 = cat8(*(const v8h*)qrow, *(const v8h*)(qrow + 16));
  v16h AQ1 = cat8(*(const v8h*)(qrow + 32), *(const v8h*)(qrow + 48));

  float rowmax[8], rowsum[8];
#pragma unroll
  for (int j = 0; j < 8; ++j) { rowmax[j] = -1e30f; rowsum[j] = 0.f; }
  v8f accO[4];
#pragma unroll
  for (int t = 0; t < 4; ++t) { v8f z = {}; accO[t] = z; }

  for (int mt = 0; mt < 128; ++mt) {
    int mb = mt * 32;
    const _Float16* k0 = k_cl + (((size_t)b * 4096) + mb + lo) * 64 + hi * 8;
    const _Float16* k1 = k0 + 16 * 64;
    if (mt + 1 < 128) {            // prefetch next K block + V lines (global_prefetch_b8)
      __builtin_prefetch(k0 + 32 * 64, 0, 1);
      __builtin_prefetch(v_t + (((size_t)b * 64) + lane * 2) * 4096 + mb + 32, 0, 1);
      __builtin_prefetch(v_t + (((size_t)b * 64) + lane * 2 + 1) * 4096 + mb + 32, 0, 1);
    }
    v8f sl = {}, sr = {};
    sl = WMMA_F16(AQ0, cat8(*(const v8h*)k0, *(const v8h*)(k0 + 16)), sl);
    sl = WMMA_F16(AQ1, cat8(*(const v8h*)(k0 + 32), *(const v8h*)(k0 + 48)), sl);
    sr = WMMA_F16(AQ0, cat8(*(const v8h*)k1, *(const v8h*)(k1 + 16)), sr);
    sr = WMMA_F16(AQ1, cat8(*(const v8h*)(k1 + 32), *(const v8h*)(k1 + 48)), sr);
#pragma unroll
    for (int j = 0; j < 8; ++j) {
      float mx = fmaxf(sl[j], sr[j]);
      for (int off = 1; off < 16; off <<= 1) mx = fmaxf(mx, __shfl_xor(mx, off, 32));
      float nm = fmaxf(rowmax[j], mx);
      float pl = __expf(sl[j] - nm), pr = __expf(sr[j] - nm);
      float ps = pl + pr;
      for (int off = 1; off < 16; off <<= 1) ps += __shfl_xor(ps, off, 32);
      float corr = __expf(rowmax[j] - nm);
      rowsum[j] = rowsum[j] * corr + ps;
      rowmax[j] = nm;
#pragma unroll
      for (int t = 0; t < 4; ++t) accO[t][j] *= corr;
      int r = (j + 8 * hi) * 32;
      sP[r + lo] = (_Float16)pl;
      sP[r + 16 + lo] = (_Float16)pr;
    }
    asm volatile("s_wait_dscnt 0x0" ::: "memory");   // P visible to own wave's ds reads
    int pb = lo * 32 + hi * 8;
    v16h AP = cat8(*(const v8h*)&sP[pb], *(const v8h*)&sP[pb + 16]);
#pragma unroll
    for (int t = 0; t < 4; ++t) {
      const _Float16* vp = v_t + (((size_t)b * 64) + t * 16 + lo) * 4096 + mb + hi * 8;
      v16h BV = cat8(*(const v8h*)vp, *(const v8h*)(vp + 16));
      accO[t] = WMMA_F16(AP, BV, accO[t]);
    }
  }

  // normalize, stage O tile (16x64 f16) for the out-projection
#pragma unroll
  for (int j = 0; j < 8; ++j) {
    float inv = 1.0f / rowsum[j];
    int r = (j + 8 * hi) * 64;
#pragma unroll
    for (int t = 0; t < 4; ++t) sP[r + t * 16 + lo] = (_Float16)(accO[t][j] * inv);
  }
  asm volatile("s_wait_dscnt 0x0" ::: "memory");
  int ob = lo * 64 + hi * 8;
  v16h AO0 = cat8(*(const v8h*)&sP[ob], *(const v8h*)&sP[ob + 16]);
  v16h AO1 = cat8(*(const v8h*)&sP[ob + 32], *(const v8h*)&sP[ob + 48]);
#pragma unroll
  for (int t = 0; t < 4; ++t) {
    const _Float16* wp = wpack_o + ((size_t)(t * 2) * 32 + lane) * 16;
    v8f a2 = {};
    a2 = WMMA_F16(AO0, *(const v16h*)wp, a2);
    a2 = WMMA_F16(AO1, *(const v16h*)(wp + 512), a2);
    int co = t * 16 + lo;
    float bb = out_b[co];
#pragma unroll
    for (int j = 0; j < 8; ++j) {
      int n = nbase + j + 8 * hi;
      float v = a2[j] + bb + hres[(((size_t)b * 4096) + n) * 64 + co];
      out_cf[(((size_t)b * 64) + co) * 4096 + n] = v;
    }
  }
}

// ---------------- launch ----------------
extern "C" void kernel_launch(void* const* d_in, const int* in_sizes, int n_in,
                              void* d_out, int out_size, void* d_ws, size_t ws_size,
                              hipStream_t stream) {
  const float* x       = (const float*)d_in[0];
  const float* temb    = (const float*)d_in[1];
  const float* gn1_g   = (const float*)d_in[2];
  const float* gn1_b   = (const float*)d_in[3];
  const float* conv1_w = (const float*)d_in[4];
  const float* conv1_b = (const float*)d_in[5];
  const float* mlp_w   = (const float*)d_in[6];
  const float* mlp_b   = (const float*)d_in[7];
  const float* gn2_g   = (const float*)d_in[8];
  const float* gn2_b   = (const float*)d_in[9];
  const float* conv2_w = (const float*)d_in[10];
  const float* conv2_b = (const float*)d_in[11];
  const float* agn_g   = (const float*)d_in[12];
  const float* agn_b   = (const float*)d_in[13];
  const float* qkv_w   = (const float*)d_in[14];
  const float* out_w   = (const float*)d_in[15];
  const float* out_bv  = (const float*)d_in[16];
  float* out = (float*)d_out;

  char* ws = (char*)d_ws;
  size_t off = 0;
  auto alloc = [&](size_t bytes) { size_t r = off; off = (off + bytes + 255) & ~(size_t)255; return r; };
  const size_t NEL = (size_t)4 * 4096 * 64;           // 1,048,576 elems
  float*    stats = (float*)   (ws + alloc(64 * 2 * sizeof(float)));
  float*    tbias = (float*)   (ws + alloc(256 * sizeof(float)));
  _Float16* act   = (_Float16*)(ws + alloc(NEL * 2));
  float*    h1    = (float*)   (ws + alloc(NEL * 4));
  float*    hres  = (float*)   (ws + alloc(NEL * 4));
  _Float16* q_cl  = (_Float16*)(ws + alloc(NEL * 2));
  _Float16* k_cl  = (_Float16*)(ws + alloc(NEL * 2));
  _Float16* v_t   = (_Float16*)(ws + alloc(NEL * 2));
  _Float16* pk1   = (_Float16*)(ws + alloc((size_t)110592 * 2));
  _Float16* pk2   = (_Float16*)(ws + alloc((size_t)110592 * 2));
  _Float16* pkq   = (_Float16*)(ws + alloc((size_t)12 * 1024 * 2));
  _Float16* pko   = (_Float16*)(ws + alloc((size_t)4 * 1024 * 2));

  // weight packing + time MLP
  pack_conv_k<<<(110592 + 255) / 256, 256, 0, stream>>>(conv1_w, pk1);
  pack_conv_k<<<(110592 + 255) / 256, 256, 0, stream>>>(conv2_w, pk2);
  pack_mat_k<<<(12 * 1024) / 256, 256, 0, stream>>>(qkv_w, pkq, 12);
  pack_mat_k<<<(4 * 1024) / 256, 256, 0, stream>>>(out_w, pko, 4);
  time_mlp_k<<<1, 256, 0, stream>>>(temb, mlp_w, mlp_b, tbias);

  // block1: GN + swish + conv1 (+ conv1_b + time bias)
  gn_stats_cf<<<64, 256, 0, stream>>>(x, stats);
  gn_apply_cf<<<(int)(NEL / 256), 256, 0, stream>>>(x, gn1_g, gn1_b, stats, act, 1);
  conv3_wmma<<<1024, 128, 0, stream>>>(act, pk1, conv1_b, tbias, nullptr, h1);

  // block2: GN + swish + conv2 (+ conv2_b + x residual)
  gn_stats_cl<<<64, 256, 0, stream>>>(h1, stats);
  gn_apply_cl<<<(int)(NEL / 256), 256, 0, stream>>>(h1, gn2_g, gn2_b, stats, act, 1);
  conv3_wmma<<<1024, 128, 0, stream>>>(act, pk2, conv2_b, nullptr, x, hres);

  // attention: GN (no swish), qkv, flash attn + out proj + residual
  gn_stats_cl<<<64, 256, 0, stream>>>(hres, stats);
  gn_apply_cl<<<(int)(NEL / 256), 256, 0, stream>>>(hres, agn_g, agn_b, stats, act, 0);
  qkv_wmma<<<1024, 128, 0, stream>>>(act, pkq, q_cl, k_cl, v_t);
  flash_wmma<<<128, 256, 0, stream>>>(q_cl, k_cl, v_t, pko, out_bv, hres, out);

  (void)in_sizes; (void)n_in; (void)out_size; (void)ws_size;
}
